// MADE_29171417874753
// MI455X (gfx1250) — matrix operand order repaired and checked by reading, not compiled, not run on hardware
//
#include <hip/hip_runtime.h>

#define MADE_D 128
#define MADE_H 512
#define MADE_B 256
#define BT 16          // batch rows per workgroup (WMMA M)
#define XST 136        // padded f16 row stride for x  (136*2 = 272B = 17*16)
#define HST 520        // padded f16 row stride for h  (520*2 = 1040B = 65*16)

typedef __attribute__((ext_vector_type(16))) _Float16 v16h;
typedef __attribute__((ext_vector_type(8)))  _Float16 v8h;
typedef __attribute__((ext_vector_type(8)))  float    v8f;

union AFrag { v16h v; v8h h[2]; };

__device__ __forceinline__ v8f wmma16(v16h a, v16h b, v8f c) {
  // D = A(16x32 f16) * B(32x16 f16) + C(16x16 f32)
  return __builtin_amdgcn_wmma_f32_16x16x32_f16(false, a, false, b,
                                                (short)0, c, false, false);
}

// Pre-mask weights (m*W), convert to f16, and store directly in the WMMA
// B-operand fragment layout: element e of lane L in (n_tile, k_chunk) is
// B[k][n] with k = kc*32 + (L/16)*8 + (e%8) + (e/8)*16, n = nt*16 + (L%16).
__global__ void made_prep_frag(const float* __restrict__ W, const float* __restrict__ M,
                               _Float16* __restrict__ dst, int Kdim, int rowOff, int total)
{
  int t = blockIdx.x * 256 + threadIdx.x;
  if (t >= total) return;
  int e    = t & 15;
  int lane = (t >> 4) & 31;
  int fi   = t >> 9;              // nt*KC + kc
  int KC   = Kdim >> 5;
  int nt   = fi / KC;
  int kc   = fi - nt * KC;
  int n = nt * 16 + (lane & 15);
  int k = kc * 32 + (lane >> 4) * 8 + (e & 7) + (e >> 3) * 16;
  int src = (rowOff + n) * Kdim + k;
  dst[t] = (_Float16)(W[src] * M[src]);
}

// Elementwise masked weights kept in f32 (for the precise per-step column dots).
__global__ void made_prep_mw(const float* __restrict__ W, const float* __restrict__ M,
                             float* __restrict__ dst, int total)
{
  int t = blockIdx.x * 256 + threadIdx.x;
  if (t < total) dst[t] = W[t] * M[t];
}

__global__ __launch_bounds__(256, 1)
void made_scan(const float* __restrict__ z,
               const float* __restrict__ b0v,
               const float* __restrict__ b1v,
               const float* __restrict__ b2v,
               const _Float16* __restrict__ M0B,
               const _Float16* __restrict__ M1B,
               const _Float16* __restrict__ M2B,
               const float* __restrict__ M2W,
               float* __restrict__ out)
{
  __shared__ float xs[BT][MADE_D];     // f32 master x
  __shared__ float zs[BT][MADE_D];     // z tile
  __shared__ __attribute__((aligned(16))) _Float16 xh[BT][XST];
  __shared__ __attribute__((aligned(16))) _Float16 h1[BT][HST];
  __shared__ __attribute__((aligned(16))) _Float16 h2[BT][HST];
  __shared__ float part[8][BT];        // per-wave logdet partials

  const int t     = threadIdx.x;
  const int lane  = t & 31;
  const int wave  = t >> 5;
  const int lmod  = lane & 15;
  const int lhalf = lane >> 4;
  const int aoff  = lhalf * 8;         // A-fragment K sub-base per lane half
  const int r0    = blockIdx.x * BT;

  for (int idx = t; idx < BT * MADE_D; idx += 256) {
    int m = idx >> 7, k = idx & (MADE_D - 1);
    xs[m][k] = 0.0f;
    zs[m][k] = z[(r0 + m) * MADE_D + k];
  }
  for (int idx = t; idx < BT * XST; idx += 256)
    (&xh[0][0])[idx] = (_Float16)0.0f;
  __syncthreads();

  for (int i = 0; i < MADE_D; ++i) {
    // ---------- layer 1: h1 = relu(x @ M0^T + b0), C is 16 x 512 ----------
    {
      v8f acc[4] = {};
      for (int kc = 0; kc < 4; ++kc) {
        AFrag a;
        const _Float16* ar = &xh[lmod][kc * 32 + aoff];
        a.h[0] = *(const v8h*)ar;          // K = kc*32 + aoff + 0..7
        a.h[1] = *(const v8h*)(ar + 16);   // K = kc*32 + aoff + 16..23
        #pragma unroll
        for (int j = 0; j < 4; ++j) {
          const int nt = wave * 4 + j;
          v16h bm = *(const v16h*)(M0B + (size_t)(((nt * 4) + kc) * 32 + lane) * 16);
          acc[j] = wmma16(a.v, bm, acc[j]);
        }
      }
      #pragma unroll
      for (int j = 0; j < 4; ++j) {
        const int n = (wave * 4 + j) * 16 + lmod;
        const float bias = b0v[n];
        #pragma unroll
        for (int r = 0; r < 8; ++r) {
          float v = acc[j][r] + bias;
          h1[r + 8 * lhalf][n] = (_Float16)(v > 0.0f ? v : 0.0f);
        }
      }
    }
    __syncthreads();
    // ---------- layer 2: h2 = relu(h1 @ M1^T + b1), C is 16 x 512 ----------
    {
      v8f acc[4] = {};
      for (int kc = 0; kc < 16; ++kc) {
        AFrag a;
        const _Float16* ar = &h1[lmod][kc * 32 + aoff];
        a.h[0] = *(const v8h*)ar;
        a.h[1] = *(const v8h*)(ar + 16);
        #pragma unroll
        for (int j = 0; j < 4; ++j) {
          const int nt = wave * 4 + j;
          v16h bm = *(const v16h*)(M1B + (size_t)(((nt * 16) + kc) * 32 + lane) * 16);
          acc[j] = wmma16(a.v, bm, acc[j]);
        }
      }
      #pragma unroll
      for (int j = 0; j < 4; ++j) {
        const int n = (wave * 4 + j) * 16 + lmod;
        const float bias = b1v[n];
        #pragma unroll
        for (int r = 0; r < 8; ++r) {
          float v = acc[j][r] + bias;
          h2[r + 8 * lhalf][n] = (_Float16)(v > 0.0f ? v : 0.0f);
        }
      }
    }
    __syncthreads();
    // ---------- only output columns i (mu) and D+i (alpha) are needed ----------
    {
      const int row = t >> 4;            // batch row 0..15
      const int sl  = t & 15;            // K slice
      const float* wm = M2W + (size_t)i * MADE_H + sl * 32;
      const float* wa = M2W + (size_t)(MADE_D + i) * MADE_H + sl * 32;
      const _Float16* hr = &h2[row][sl * 32];
      float smu = 0.0f, sal = 0.0f;
      #pragma unroll
      for (int j2 = 0; j2 < 32; ++j2) {
        float hv = (float)hr[j2];
        smu += hv * wm[j2];
        sal += hv * wa[j2];
      }
      #pragma unroll
      for (int off = 8; off >= 1; off >>= 1) {  // butterfly within 16-lane group
        smu += __shfl_xor(smu, off, 32);
        sal += __shfl_xor(sal, off, 32);
      }
      if (sl == 0) {
        float mu = smu + b2v[i];
        float al = sal + b2v[MADE_D + i];
        float xi = mu + __expf(al) * zs[row][i];
        xs[row][i] = xi;
        xh[row][i] = (_Float16)xi;
      }
    }
    __syncthreads();
  }

  // ---------- full alpha (16x128) from last-step h2, then log_det ----------
  {
    const int nt = wave;                 // 8 waves, 8 N-tiles of 16
    v8f acc = {};
    for (int kc = 0; kc < 16; ++kc) {
      AFrag a;
      const _Float16* ar = &h2[lmod][kc * 32 + aoff];
      a.h[0] = *(const v8h*)ar;
      a.h[1] = *(const v8h*)(ar + 16);
      v16h bm = *(const v16h*)(M2B + (size_t)(((nt * 16) + kc) * 32 + lane) * 16);
      acc = wmma16(a.v, bm, acc);
    }
    const float bias = b2v[MADE_D + nt * 16 + lmod];
    #pragma unroll
    for (int r = 0; r < 8; ++r) {
      float v = acc[r] + bias;
      #pragma unroll
      for (int off = 8; off >= 1; off >>= 1)
        v += __shfl_xor(v, off, 32);
      if (lmod == 0) part[wave][r + 8 * lhalf] = v;   // deterministic partials
    }
  }
  __syncthreads();

  for (int idx = t; idx < BT * MADE_D; idx += 256) {
    int m = idx >> 7, k = idx & (MADE_D - 1);
    out[(size_t)(r0 + m) * MADE_D + k] = xs[m][k];
  }
  if (t < BT) {
    float s = 0.0f;
    #pragma unroll
    for (int w = 0; w < 8; ++w) s += part[w][t];     // fixed-order reduction
    out[MADE_B * MADE_D + r0 + t] = -s;
  }
}

extern "C" void kernel_launch(void* const* d_in, const int* in_sizes, int n_in,
                              void* d_out, int out_size, void* d_ws, size_t ws_size,
                              hipStream_t stream)
{
  const float* z  = (const float*)d_in[0];
  const float* W0 = (const float*)d_in[1];
  const float* b0 = (const float*)d_in[2];
  const float* W1 = (const float*)d_in[3];
  const float* b1 = (const float*)d_in[4];
  const float* W2 = (const float*)d_in[5];
  const float* b2 = (const float*)d_in[6];
  const float* m0 = (const float*)d_in[7];
  const float* m1 = (const float*)d_in[8];
  const float* m2 = (const float*)d_in[9];
  float* out = (float*)d_out;

  // Workspace layout (1.25 MB total):
  char* ws = (char*)d_ws;
  _Float16* M0B = (_Float16*)(ws);                      // 512*128 f16 = 128KB
  _Float16* M1B = (_Float16*)(ws + (size_t)128 * 1024); // 512*512 f16 = 512KB
  _Float16* M2B = (_Float16*)(ws + (size_t)640 * 1024); // 128*512 f16 = 128KB (alpha rows)
  float*    M2W = (float*)   (ws + (size_t)768 * 1024); // 256*512 f32 = 512KB

  { int total = MADE_H * MADE_D;   // layer-1 B fragments
    made_prep_frag<<<(total + 255) / 256, 256, 0, stream>>>(W0, m0, M0B, MADE_D, 0, total); }
  { int total = MADE_H * MADE_H;   // layer-2 B fragments
    made_prep_frag<<<(total + 255) / 256, 256, 0, stream>>>(W1, m1, M1B, MADE_H, 0, total); }
  { int total = MADE_D * MADE_H;   // final alpha B fragments (rows D..2D-1 of W2)
    made_prep_frag<<<(total + 255) / 256, 256, 0, stream>>>(W2, m2, M2B, MADE_H, MADE_D, total); }
  { int total = 2 * MADE_D * MADE_H;
    made_prep_mw<<<(total + 255) / 256, 256, 0, stream>>>(W2, m2, M2W, total); }

  made_scan<<<MADE_B / BT, 256, 0, stream>>>(z, b0, b1, b2, M0B, M1B, M2B, M2W, out);
}
